// MoCoEncoder_34780645163707
// MI455X (gfx1250) — compile-verified, matrix operand order checked by reading, not run on hardware
//
#include <hip/hip_runtime.h>

// ---- problem constants (match reference) ----
#define Bn   4
#define Cc   128
#define EMB  64
#define KK   512
#define SS   32768              // D*H*W = 32*32*32
#define NN   (Bn * SS)          // 131072 voxels
#define TILE 128                // voxels per block
#define Q_ELEMS (NN * EMB)      // 8388608 floats of quantize output

typedef __attribute__((ext_vector_type(16))) _Float16 v16h;
typedef __attribute__((ext_vector_type(8)))  float    v8f;

union FragH { v16h h; unsigned u[8]; };

// 16-bit A/B fragment K mapping (ISA 7.12.2): dword j of the lane's 8 dwords
// holds K = base(j) + 8*half (pair K, K+1). base = 2j for j<4, 2j+8 for j>=4.
__device__ __forceinline__ int kpair(int j, int half) {
  return ((j < 4) ? (2 * j) : (2 * j + 8)) + 8 * half;
}

// ---------------- init: e-norms, f16 codebook, zero diff slot ----------------
__global__ void vq_init(const float* __restrict__ embed,
                        float* __restrict__ ws_en,
                        _Float16* __restrict__ embed_h,
                        float* __restrict__ diff_slot) {
  const int t = threadIdx.x;
  if (t == 0) *diff_slot = 0.0f;
  for (int k = t; k < KK; k += 256) {
    float s = 0.0f;
    for (int e = 0; e < EMB; ++e) { float v = embed[e * KK + k]; s += v * v; }
    ws_en[k] = s;
  }
  for (int i = t; i < EMB * KK; i += 256) embed_h[i] = (_Float16)embed[i];
}

// ---------------- fused conv-GEMM + codebook distance argmin ----------------
__global__ __launch_bounds__(256) void vq_fused(
    const float* __restrict__ x,        // (B,C,S) channel-major
    const float* __restrict__ conv_w,   // (EMB,C)
    const float* __restrict__ conv_b,   // (EMB)
    const float* __restrict__ embed,    // (EMB,K) fp32
    const float* __restrict__ ws_en,    // (K) ||e_k||^2
    const _Float16* __restrict__ embed_h, // (EMB,K) f16
    float* __restrict__ out_q,          // (B,EMB,S)
    float* __restrict__ diff_slot,
    int* __restrict__ out_ind) {
  __shared__ _Float16 wsB[EMB * 130];   // conv_w^T: [o][c], pitch 130 (16640B)
  __shared__ _Float16 zb[TILE * 66];    // z tile f16: [voxel][e]      (16896B)
  __shared__ _Float16 eb[EMB * 66];     // codebook chunk: [code][e]   (8448B)
  __shared__ float    enl[KK];          // ||e_k||^2                   (2048B)
  __shared__ int      indl[TILE];       //                              (512B)

  const int t    = threadIdx.x;
  const int lane = t & 31;
  const int wave = t >> 5;
  const int half = lane >> 4;
  const int l16  = lane & 15;

  const int n0 = blockIdx.x * TILE;     // tile never crosses a batch (S%128==0)
  const int b  = n0 >> 15;              // / SS
  const int s0 = n0 & (SS - 1);

  // stage conv_w transposed (coalesced read, f16 convert) and e-norms
  for (int i = t; i < EMB * Cc; i += 256) {
    int o = i >> 7, c = i & 127;
    wsB[o * 130 + c] = (_Float16)conv_w[i];
  }
  for (int k = t; k < KK; k += 256) enl[k] = ws_en[k];
  __syncthreads();

  // ================= GEMM1: z[16 x 64] per wave =================
  const int m0 = wave * 16;
  const float* xb = x + (size_t)b * Cc * SS;
  const size_t sv = (size_t)(s0 + m0 + l16);

  v8f acc[4] = {};
  for (int kt = 0; kt < 4; ++kt) {
    const int c0 = kt * 32;
    FragH a;
#pragma unroll
    for (int j = 0; j < 8; ++j) {
      int c = c0 + kpair(j, half);
      // lanes 0..15 read 16 consecutive floats per channel: coalesced 64B
      a.h[2 * j]     = (_Float16)__builtin_nontemporal_load(&xb[(size_t)c * SS + sv]);
      a.h[2 * j + 1] = (_Float16)__builtin_nontemporal_load(&xb[(size_t)(c + 1) * SS + sv]);
    }
#pragma unroll
    for (int ot = 0; ot < 4; ++ot) {
      FragH bf;
      const int o = ot * 16 + l16;
#pragma unroll
      for (int j = 0; j < 8; ++j)
        bf.u[j] = *(const unsigned*)&wsB[o * 130 + c0 + kpair(j, half)];
      acc[ot] = __builtin_amdgcn_wmma_f32_16x16x32_f16(
          false, a.h, false, bf.h, (short)0, acc[ot], false, false);
    }
  }

  // z (+bias) -> LDS f16, per C/D layout: VGPR r, lane -> (M=r+8*half, N=l16)
#pragma unroll
  for (int ot = 0; ot < 4; ++ot) {
    const int o = ot * 16 + l16;
    const float cb = conv_b[o];
#pragma unroll
    for (int r = 0; r < 8; ++r) {
      int m = r + 8 * half;
      zb[(m0 + m) * 66 + o] = (_Float16)(acc[ot][r] + cb);
    }
  }
  __syncthreads();

  // ================= GEMM2: -2 f.e + ||e||^2, argmin over K ================
  FragH a0, a1;                        // chunk-invariant A fragments (e = EMB dim)
#pragma unroll
  for (int j = 0; j < 8; ++j) {
    int e = kpair(j, half);
    a0.u[j] = *(const unsigned*)&zb[(m0 + l16) * 66 + e];
    a1.u[j] = *(const unsigned*)&zb[(m0 + l16) * 66 + e + 32];
  }

  float minv[8];
  int   mini[8];
#pragma unroll
  for (int r = 0; r < 8; ++r) { minv[r] = 3.4e38f; mini[r] = 0; }

  for (int kc = 0; kc < 8; ++kc) {     // 8 chunks of 64 codes
    __syncthreads();
    for (int i = t; i < EMB * EMB; i += 256) {   // stage chunk, transposed
      int e = i >> 6, kk = i & 63;
      eb[kk * 66 + e] = embed_h[e * KK + kc * 64 + kk];
    }
    __syncthreads();
#pragma unroll
    for (int ct = 0; ct < 4; ++ct) {
      FragH b0, b1;
      const int kcol = ct * 16 + l16;
#pragma unroll
      for (int j = 0; j < 8; ++j) {
        int e = kpair(j, half);
        b0.u[j] = *(const unsigned*)&eb[kcol * 66 + e];
        b1.u[j] = *(const unsigned*)&eb[kcol * 66 + e + 32];
      }
      v8f sc = {};
      sc = __builtin_amdgcn_wmma_f32_16x16x32_f16(false, a0.h, false, b0.h,
                                                  (short)0, sc, false, false);
      sc = __builtin_amdgcn_wmma_f32_16x16x32_f16(false, a1.h, false, b1.h,
                                                  (short)0, sc, false, false);
      const int kglob = kc * 64 + ct * 16 + l16;
      const float en = enl[kglob];
#pragma unroll
      for (int r = 0; r < 8; ++r) {
        float d = en - 2.0f * sc[r];
        if (d < minv[r]) { minv[r] = d; mini[r] = kglob; }
      }
    }
  }

  // butterfly min-reduce across the 16 lanes holding the same voxel row
#pragma unroll
  for (int r = 0; r < 8; ++r) {
    float v = minv[r];
    int idx = mini[r];
#pragma unroll
    for (int off = 1; off < 16; off <<= 1) {
      float ov = __shfl_xor(v, off, 32);
      int   oi = __shfl_xor(idx, off, 32);
      if (ov < v || (ov == v && oi < idx)) { v = ov; idx = oi; }  // first-min tiebreak
    }
    if (l16 == 0) indl[m0 + r + 8 * half] = idx;
  }
  __syncthreads();

  // ================= outputs: gather, transpose-store, diff =================
  const int vx = t & 127;
  const int e0 = (t >> 7) * 32;
  const int n  = n0 + vx;
  const int ind = indl[vx];
  if (t < TILE) out_ind[n] = ind;

  float lsum = 0.0f;
  const size_t obase = (size_t)b * EMB * SS + (size_t)(s0 + vx);
#pragma unroll 4
  for (int e = e0; e < e0 + 32; ++e) {
    float q  = embed[e * KK + ind];            // fp32 codebook, L2-resident
    float zz = (float)zb[vx * 66 + e];
    float dq = q - zz;
    lsum += dq * dq;
    // threads 0..127 store 128 contiguous floats per plane: fully coalesced
    __builtin_nontemporal_store(q, &out_q[obase + (size_t)e * SS]);
  }
#pragma unroll
  for (int off = 1; off < 32; off <<= 1) lsum += __shfl_xor(lsum, off, 32);
  if (lane == 0)
    atomicAdd(diff_slot, lsum * (1.0f / ((float)NN * (float)EMB)));
}

extern "C" void kernel_launch(void* const* d_in, const int* in_sizes, int n_in,
                              void* d_out, int out_size, void* d_ws, size_t ws_size,
                              hipStream_t stream) {
  const float* x      = (const float*)d_in[0];
  const float* conv_w = (const float*)d_in[1];
  const float* conv_b = (const float*)d_in[2];
  const float* embed  = (const float*)d_in[3];

  float* outf      = (float*)d_out;
  float* diff_slot = outf + Q_ELEMS;          // layout: quantize | diff | indices
  int*   out_ind   = (int*)(outf + Q_ELEMS + 1);

  float*     ws_en   = (float*)d_ws;          // K floats
  _Float16*  embed_h = (_Float16*)(ws_en + KK); // EMB*K f16

  vq_init<<<1, 256, 0, stream>>>(embed, ws_en, embed_h, diff_slot);
  vq_fused<<<NN / TILE, 256, 0, stream>>>(x, conv_w, conv_b, embed,
                                          ws_en, embed_h, outf, diff_slot, out_ind);
}
